// VectorQuantizer_52690658788133
// MI455X (gfx1250) — matrix operand order repaired and checked by reading, not compiled, not run on hardware
//
#include <hip/hip_runtime.h>
#include <hip/hip_bf16.h>
#include <float.h>

typedef __attribute__((ext_vector_type(16))) __bf16 v16bf;
typedef __attribute__((ext_vector_type(8)))  float  v8f;
typedef __attribute__((ext_vector_type(4)))  unsigned int u32x4;
typedef __attribute__((ext_vector_type(8)))  int  i32x8;
typedef __attribute__((ext_vector_type(4)))  int  i32x4;

namespace {
constexpr int Bn  = 32;
constexpr int Cn  = 256;
constexpr int Hn  = 64;
constexpr int Wn  = 64;
constexpr int Kn  = 1024;
constexpr int HW  = Hn * Wn;          // 4096
constexpr int CHW = Cn * HW;          // 1048576
constexpr int Nn  = Bn * HW;          // 131072
constexpr float COMMIT = 0.25f;
constexpr int NKT = Kn / 16;          // 64 column tiles
constexpr int PANEL_BF16 = 4096;      // 8KB B panel per column tile
}

// -------- kernel 1: e_norms[k] = ||emb[k]||^2  (one wave32 per codeword) -----
__global__ void vq_enorm_kernel(const float* __restrict__ emb,
                                float* __restrict__ enorm) {
  const int wave = (blockIdx.x * blockDim.x + threadIdx.x) >> 5;
  const int lane = threadIdx.x & 31;
  const float4* row = reinterpret_cast<const float4*>(emb + (size_t)wave * Cn);
  const float4 a = row[lane];        // elems 0..127
  const float4 b = row[lane + 32];   // elems 128..255
  float s = a.x*a.x + a.y*a.y + a.z*a.z + a.w*a.w
          + b.x*b.x + b.y*b.y + b.z*b.z + b.w*b.w;
#pragma unroll
  for (int m = 16; m >= 1; m >>= 1) s += __shfl_xor(s, m, 32);
  if (lane == 0) enorm[wave] = s;
}

// -------- kernel 2: pre-swizzle emb -> bf16 WMMA B-fragment layout ----------
// bswz[((kt*8+ks)*32 + lane)*16 + e] = bf16( emb[ col=kt*16+(lane&15),
//                                          c = ks*32 + (lane<16 ? e : 16+e) ] )
// One column tile kt = one contiguous 8KB panel (TDM-friendly 1-D tile).
__global__ void vq_bswz_kernel(const float* __restrict__ emb,
                               __bf16* __restrict__ bswz) {
  const int tid  = blockIdx.x * blockDim.x + threadIdx.x; // 0 .. K*C-1
  const int e    = tid & 15;
  const int lane = (tid >> 4) & 31;
  const int ksl  = tid >> 9;        // kt*8 + ks
  const int kt   = ksl >> 3;
  const int ks   = ksl & 7;
  const int col  = kt * 16 + (lane & 15);
  const int c    = ks * 32 + ((lane < 16) ? e : (16 + e));
  bswz[tid] = (__bf16)emb[(size_t)col * Cn + c];
}

// ---- TDM: DMA one contiguous 8KB panel (1024 x 8B) from global into LDS ----
#if __has_builtin(__builtin_amdgcn_tensor_load_to_lds)
#define VQ_HAVE_TDM 1
__device__ __forceinline__ void tdm_load_panel(const void* gsrc, void* ldst) {
  const unsigned long long ga = (unsigned long long)(uintptr_t)gsrc;
  const unsigned int la = (unsigned int)(uintptr_t)ldst; // low 32 = LDS offset
  u32x4 g0;
  g0[0] = 1u;                                           // count=1, user desc
  g0[1] = la;                                           // lds_addr
  g0[2] = (unsigned int)ga;                             // global_addr[31:0]
  g0[3] = (unsigned int)((ga >> 32) & 0x1FFFFFFu)       // global_addr[56:32]
        | (2u << 30);                                   // type = 2 ("image")
  i32x8 g1;
  g1[0] = (int)(3u << 16);      // workgroup_mask=0, data_size=3 (8B), no flags
  g1[1] = (int)(1024u << 16);   // atomic_barrier_addr=0 | tensor_dim0.lo16=1024
  g1[2] = (int)(1u << 16);      // tensor_dim0.hi16=0    | tensor_dim1.lo16=1
  g1[3] = (int)(1024u << 16);   // tensor_dim1.hi16=0    | tile_dim0=1024
  g1[4] = 1;                    // tile_dim1=1, tile_dim2=0
  g1[5] = 1024;                 // tensor_dim0_stride.lo32 = 1024
  g1[6] = 0;                    // stride hi16 | tensor_dim1_stride.lo16
  g1[7] = 0;
  const i32x4 z4 = {0, 0, 0, 0};              // groups 2/3 unused (<=2D tensor)
  const i32x8 z8 = {0, 0, 0, 0, 0, 0, 0, 0};  // trailing group (6-arg variant)
  __builtin_amdgcn_tensor_load_to_lds(g0, g1, z4, z4, z8, 0);
}
#else
#define VQ_HAVE_TDM 0
#endif

// -------- kernel 3: WMMA bf16 scores + fused argmin over K ------------------
// block = 256 threads (8 waves); wave owns rows [n0, n0+32): two 16-row tiles.
// The 8KB B panel for each column tile is shared by all 8 waves: wave 0 DMAs
// panel kt+1 into the ping-pong LDS buffer via the Tensor Data Mover while the
// block runs 16 WMMAs against panel kt out of LDS.
__global__ void __launch_bounds__(256, 1)
vq_argmin_kernel(const float* __restrict__ x, const __bf16* __restrict__ bswz,
                 const float* __restrict__ enorm, int* __restrict__ idx) {
  __shared__ __align__(32) __bf16 ldsb[2][PANEL_BF16];   // 2 x 8KB

  const int lane = threadIdx.x & 31;
  const int wave = threadIdx.x >> 5;
  const int n0   = blockIdx.x * 256 + wave * 32;
  const int b    = n0 >> 12;          // n0 / 4096 (rows never cross a batch)
  const int hw0  = n0 & 4095;
  const int mrow = lane & 15;
  const bool hi  = lane >= 16;

#if VQ_HAVE_TDM
  if (wave == 0) tdm_load_panel(bswz, &ldsb[0][0]);      // panel 0 in flight
#else
  {
    const float4* s = reinterpret_cast<const float4*>(bswz);
    float4* d = reinterpret_cast<float4*>(&ldsb[0][0]);
    d[threadIdx.x]       = s[threadIdx.x];
    d[threadIdx.x + 256] = s[threadIdx.x + 256];
  }
#endif

  const float* xb = x + (size_t)b * CHW + hw0 + mrow;

  // Build A fragments (contiguous k order, same as the bswz B layout) — overlaps the DMA.
  // lane<16:  elems 0..15 -> K 0..15
  // lane>=16: elems 0..15 -> K 16..31
  v16bf afrag[2][8];
#pragma unroll
  for (int ks = 0; ks < 8; ++ks) {
#pragma unroll
    for (int e = 0; e < 16; ++e) {
      const int c = ks * 32 + (hi ? (e + 16) : e);
      afrag[0][ks][e] = (__bf16)xb[(size_t)c * HW];        // rows n0..n0+15
      afrag[1][ks][e] = (__bf16)xb[(size_t)c * HW + 16];   // rows n0+16..n0+31
    }
  }

  float minv[2][8];
  int   mini[2][8];
#pragma unroll
  for (int t = 0; t < 2; ++t)
#pragma unroll
    for (int r = 0; r < 8; ++r) { minv[t][r] = FLT_MAX; mini[t][r] = 0; }

#if VQ_HAVE_TDM
  if (wave == 0) __builtin_amdgcn_s_wait_tensorcnt(0);
#endif
  __syncthreads();                                       // panel 0 ready

  for (int kt = 0; kt < NKT; ++kt) {
#if VQ_HAVE_TDM
    if (wave == 0 && kt + 1 < NKT)                       // DMA next panel now
      tdm_load_panel(bswz + (size_t)(kt + 1) * PANEL_BF16,
                     &ldsb[(kt + 1) & 1][0]);
#else
    if (kt + 1 < NKT) {
      const float4* s = reinterpret_cast<const float4*>(
          bswz + (size_t)(kt + 1) * PANEL_BF16);
      float4* d = reinterpret_cast<float4*>(&ldsb[(kt + 1) & 1][0]);
      d[threadIdx.x]       = s[threadIdx.x];
      d[threadIdx.x + 256] = s[threadIdx.x + 256];
    }
#endif
    v8f acc0 = {};
    v8f acc1 = {};
    const float en = enorm[kt * 16 + mrow];
    const __bf16* bp = &ldsb[kt & 1][(size_t)lane << 4];
#pragma unroll
    for (int ks = 0; ks < 8; ++ks) {
      const v16bf bfrag =
          *reinterpret_cast<const v16bf*>(bp + ((size_t)ks << 9));
      acc0 = __builtin_amdgcn_wmma_f32_16x16x32_bf16(
          false, afrag[0][ks], false, bfrag, (short)0, acc0, false, false);
      acc1 = __builtin_amdgcn_wmma_f32_16x16x32_bf16(
          false, afrag[1][ks], false, bfrag, (short)0, acc1, false, false);
    }
#pragma unroll
    for (int r = 0; r < 8; ++r) {
      const int col = kt * 16 + mrow;          // N index held by this lane
      const float d0 = en - 2.0f * acc0[r];
      if (d0 < minv[0][r]) { minv[0][r] = d0; mini[0][r] = col; }
      const float d1 = en - 2.0f * acc1[r];
      if (d1 < minv[1][r]) { minv[1][r] = d1; mini[1][r] = col; }
    }
#if VQ_HAVE_TDM
    if (wave == 0) __builtin_amdgcn_s_wait_tensorcnt(0); // next panel landed
#endif
    __syncthreads();     // drains DS reads of this panel; publishes next panel
  }

  // Cross-lane argmin within each 16-lane group (all N columns of one row m).
#pragma unroll
  for (int t = 0; t < 2; ++t) {
#pragma unroll
    for (int r = 0; r < 8; ++r) {
      float v  = minv[t][r];
      int   ii = mini[t][r];
#pragma unroll
      for (int m = 8; m >= 1; m >>= 1) {
        const float ov = __shfl_xor(v, m, 32);
        const int   oi = __shfl_xor(ii, m, 32);
        if (ov < v || (ov == v && oi < ii)) { v = ov; ii = oi; }
      }
      if (mrow == 0)
        idx[n0 + t * 16 + r + (hi ? 8 : 0)] = ii;  // lanes<16: m=r; lanes>=16: m=8+r
    }
  }
}

// -------- kernel 4: gather codewords, write NCHW output, partial MSE --------
__global__ void vq_gather_kernel(const float* __restrict__ x,
                                 const float* __restrict__ emb,
                                 const int* __restrict__ idx,
                                 float* __restrict__ out,
                                 float* __restrict__ partial) {
  const int n  = blockIdx.x * 256 + threadIdx.x;
  const int b  = n >> 12;
  const int hw = n & 4095;
  const float4* erow4 =
      reinterpret_cast<const float4*>(emb + (size_t)idx[n] * Cn);  // L2-resident
  const float* xin = x   + (size_t)b * CHW + hw;
  float*       o   = out + (size_t)b * CHW + hw;
  float s = 0.f;
#pragma unroll 4
  for (int c4 = 0; c4 < Cn / 4; ++c4) {
    const float4 q4 = erow4[c4];
    const int c = c4 * 4;
    const float q[4] = {q4.x, q4.y, q4.z, q4.w};
#pragma unroll
    for (int j = 0; j < 4; ++j) {
      const float xv = xin[(size_t)(c + j) * HW];
      o[(size_t)(c + j) * HW] = q[j];            // coalesced across hw
      const float d = q[j] - xv;
      s += d * d;
    }
  }
  __shared__ float red[256];
  red[threadIdx.x] = s;
  __syncthreads();
  for (int off = 128; off >= 1; off >>= 1) {
    if (threadIdx.x < off) red[threadIdx.x] += red[threadIdx.x + off];
    __syncthreads();
  }
  if (threadIdx.x == 0) partial[blockIdx.x] = red[0];
}

// -------- kernel 5: final deterministic loss reduction ----------------------
__global__ void vq_loss_kernel(const float* __restrict__ partial,
                               float* __restrict__ loss_out) {
  __shared__ float red[256];
  red[threadIdx.x] = partial[threadIdx.x] + partial[threadIdx.x + 256];
  __syncthreads();
  for (int off = 128; off >= 1; off >>= 1) {
    if (threadIdx.x < off) red[threadIdx.x] += red[threadIdx.x + off];
    __syncthreads();
  }
  if (threadIdx.x == 0) {
    const float mse = red[0] / (float)((size_t)Bn * CHW);
    loss_out[0] = (1.0f + COMMIT) * mse;   // q_latent + commit * e_latent (equal values)
  }
}

extern "C" void kernel_launch(void* const* d_in, const int* in_sizes, int n_in,
                              void* d_out, int out_size, void* d_ws, size_t ws_size,
                              hipStream_t stream) {
  (void)in_sizes; (void)n_in; (void)out_size; (void)ws_size;
  const float* x   = (const float*)d_in[0];   // inputs [B,C,H,W] f32
  const float* emb = (const float*)d_in[1];   // emb_weight [K,C] f32
  float* out = (float*)d_out;                 // [B*C*H*W] quantized, then [1] loss

  char* ws = (char*)d_ws;
  int*    idx     = (int*)ws;                                               // N * 4 B
  __bf16* bswz    = (__bf16*)(ws + (size_t)Nn * 4);                         // K*C * 2 B
  float*  enorm   = (float*)(ws + (size_t)Nn * 4 + (size_t)Kn * Cn * 2);    // K * 4 B
  float*  partial = (float*)(ws + (size_t)Nn * 4 + (size_t)Kn * Cn * 2
                                + (size_t)Kn * 4);                          // 512 * 4 B

  vq_enorm_kernel <<<Kn / 4,          128, 0, stream>>>(emb, enorm);
  vq_bswz_kernel  <<<(Kn * Cn) / 256, 256, 0, stream>>>(emb, bswz);
  vq_argmin_kernel<<<Nn / 256,        256, 0, stream>>>(x, bswz, enorm, idx);
  vq_gather_kernel<<<Nn / 256,        256, 0, stream>>>(x, emb, idx, out, partial);
  vq_loss_kernel  <<<1,               256, 0, stream>>>(partial, out + (size_t)Bn * CHW);
}
